// MLPEncoder_21638045237571
// MI455X (gfx1250) — matrix operand order repaired and checked by reading, not compiled, not run on hardware
//
#include <hip/hip_runtime.h>

// ---------------------------------------------------------------------------
// Problem constants (from the reference)
// ---------------------------------------------------------------------------
#define NV_VOCAB   50000
#define E_DIM      300
#define E_PAD      320          // E padded to multiple of 32 (zeros)
#define L_TOK      64
#define N_SENT     8192         // B * BAG
#define MD_DIM     1024
#define K1_PAD     1344         // E_PAD + MD_DIM  (42 * 32)
#define H2_DIM     512          // 2*H, GEMM1 N and GEMM2 K (16 * 32)
#define H_DIM      256          // GEMM2 N

typedef __bf16 bf16_t;
typedef __attribute__((ext_vector_type(8)))  __bf16 bf16x8;
typedef __attribute__((ext_vector_type(16))) __bf16 v16bf;
typedef __attribute__((ext_vector_type(8)))  float  v8f;

// ---------------------------------------------------------------------------
// Stage 0a: pack w1 [512,1324] f32 -> W1b [512,1344] bf16, zero pad cols 300..319
// ---------------------------------------------------------------------------
__global__ __launch_bounds__(256) void cvt_w1_kernel(const float* __restrict__ w1,
                                                     bf16_t* __restrict__ W1b) {
    int idx = blockIdx.x * 256 + threadIdx.x;
    if (idx >= H2_DIM * K1_PAD) return;
    int r = idx / K1_PAD;
    int c = idx - r * K1_PAD;
    float v = 0.0f;
    if (c < E_DIM)           v = w1[(size_t)r * (E_DIM + MD_DIM) + c];
    else if (c >= E_PAD)     v = w1[(size_t)r * (E_DIM + MD_DIM) + (c - (E_PAD - E_DIM))];
    W1b[idx] = (bf16_t)v;
}

// Stage 0b: pack w2 [256,512] f32 -> bf16 (K already multiple of 32)
__global__ __launch_bounds__(256) void cvt_w2_kernel(const float* __restrict__ w2,
                                                     bf16_t* __restrict__ W2b) {
    int idx = blockIdx.x * 256 + threadIdx.x;
    if (idx >= H_DIM * H2_DIM) return;
    W2b[idx] = (bf16_t)w2[idx];
}

// ---------------------------------------------------------------------------
// Stage 1: fused embedding-bag (gather + masked mean) + concat + bf16 pack.
// One 256-thread block (8 waves) per sentence. Gather rows live in L2 (60 MB
// table < 192 MB L2), token ids staged through LDS once per block.
// X row layout: [0,300) mean-embedding, [300,320) zeros, [320,1344) mention.
// ---------------------------------------------------------------------------
__global__ __launch_bounds__(256) void embed_concat_kernel(
        const float* __restrict__ emb, const int* __restrict__ tokens,
        const int* __restrict__ lengths, const float* __restrict__ mention,
        bf16_t* __restrict__ X) {
    __shared__ int toks[L_TOK];
    const int n = blockIdx.x;
    const int t = threadIdx.x;
    if (t < L_TOK) toks[t] = tokens[(size_t)n * L_TOK + t];
    __syncthreads();

    const int len = lengths[n];            // uniform within the block
    const float inv = 1.0f / (float)len;
    bf16_t* xrow = X + (size_t)n * K1_PAD;

    for (int e = t; e < E_PAD; e += 256) { // 320 cols -> 2 iters (tail masked)
        float s = 0.0f;
        if (e < E_DIM) {
            for (int l = 0; l < len; ++l)
                s += emb[(size_t)toks[l] * E_DIM + e];
        }
        xrow[e] = (bf16_t)(s * inv);       // cols [300,320) get exact zeros
    }
    const float* mrow = mention + (size_t)n * MD_DIM;
    for (int m = t; m < MD_DIM; m += 256)
        xrow[E_PAD + m] = (bf16_t)mrow[m];
}

// ---------------------------------------------------------------------------
// Stage 2/3: WMMA bf16 GEMM, D = relu(A[M,K] @ B[N,K]^T + bias), row-major.
// Wave tile = 32x64 output (2x4 WMMA tiles), K-step 32: 8 v_wmma per 12 b128.
// K and N are template constants so every tile stride (tn*16*K, 16*K) is a
// compile-time immediate: all 12 loads per K-step issue off just TWO base
// address VGPR pairs using the signed-24-bit VMEM IOFFSET (max 129,024 B),
// with zero per-iteration address VALU. Distinct destination registers per B
// tile let the scheduler keep many loads in flight (loadcnt waits > 0)
// instead of round 2's serialized load->wait(0)->wmma chain.
// VGPR layouts (wave32); A/B in this code's contiguous k order, C/D per CDNA5 ISA 7.12.2:
//   A 16x32 bf16 : lane (m | hi<<4) -> 16 contiguous bf16 of A row m @ k0+hi*16
//   B 32x16 bf16 : lane (n | hi<<4) -> 16 contiguous bf16 of W row n @ k0+hi*16
//   C/D 16x16 f32: VGPR r -> M = r + 8*hi, lane&15 -> N
// ---------------------------------------------------------------------------
template<int K, int N, bool OUT_BF16>
__global__ __launch_bounds__(256) void gemm_bias_relu_wmma(
        const bf16_t* __restrict__ A, const bf16_t* __restrict__ B,
        const float* __restrict__ bias, void* __restrict__ out) {
    const int lane   = threadIdx.x & 31;
    const int waveId = blockIdx.x * 8 + (threadIdx.x >> 5);
    constexpr int tilesN = N >> 6;             // 64-wide wave tiles in N
    const int wm     = (waveId / tilesN) << 5; // 32 rows
    const int wn     = (waveId % tilesN) << 6; // 64 cols
    const int l16    = lane & 15;
    const int hi     = lane >> 4;

    // Two base pointers; every other operand address is base + immediate.
    const bf16_t* ap = A + (size_t)(wm + l16) * K + hi * 16;
    const bf16_t* bp = B + (size_t)(wn + l16) * K + hi * 16;

    v8f acc[2][4] = {};

    for (int k0 = 0; k0 < K; k0 += 32) {
        // speculative prefetch 2 K-steps ahead (global_prefetch_b8)
        __builtin_prefetch(ap + k0 + 64, 0, 1);
        __builtin_prefetch(ap + 16 * K + k0 + 64, 0, 1);

        union { v16bf v; bf16x8 h[2]; } a0, a1;
        a0.h[0] = *(const bf16x8*)(ap + k0);
        a0.h[1] = *(const bf16x8*)(ap + k0 + 8);
        a1.h[0] = *(const bf16x8*)(ap + 16 * K + k0);
        a1.h[1] = *(const bf16x8*)(ap + 16 * K + k0 + 8);

        v16bf b0 = *(const v16bf*)(bp + k0);            // +      0
        v16bf b1 = *(const v16bf*)(bp + 16 * K + k0);   // + 32K  B
        v16bf b2 = *(const v16bf*)(bp + 32 * K + k0);   // + 64K  B
        v16bf b3 = *(const v16bf*)(bp + 48 * K + k0);   // + 96K  B

        acc[0][0] = __builtin_amdgcn_wmma_f32_16x16x32_bf16(
            false, a0.v, false, b0, (short)0, acc[0][0], false, false);
        acc[1][0] = __builtin_amdgcn_wmma_f32_16x16x32_bf16(
            false, a1.v, false, b0, (short)0, acc[1][0], false, false);
        acc[0][1] = __builtin_amdgcn_wmma_f32_16x16x32_bf16(
            false, a0.v, false, b1, (short)0, acc[0][1], false, false);
        acc[1][1] = __builtin_amdgcn_wmma_f32_16x16x32_bf16(
            false, a1.v, false, b1, (short)0, acc[1][1], false, false);
        acc[0][2] = __builtin_amdgcn_wmma_f32_16x16x32_bf16(
            false, a0.v, false, b2, (short)0, acc[0][2], false, false);
        acc[1][2] = __builtin_amdgcn_wmma_f32_16x16x32_bf16(
            false, a1.v, false, b2, (short)0, acc[1][2], false, false);
        acc[0][3] = __builtin_amdgcn_wmma_f32_16x16x32_bf16(
            false, a0.v, false, b3, (short)0, acc[0][3], false, false);
        acc[1][3] = __builtin_amdgcn_wmma_f32_16x16x32_bf16(
            false, a1.v, false, b3, (short)0, acc[1][3], false, false);
    }

    float*  outf = (float*)out;
    bf16_t* outb = (bf16_t*)out;

#pragma unroll
    for (int tn = 0; tn < 4; ++tn) {
        const int col = wn + tn * 16 + l16;
        const float bc = bias[col];
#pragma unroll
        for (int tm = 0; tm < 2; ++tm) {
            const int rowBase = wm + tm * 16 + 8 * hi;
#pragma unroll
            for (int r = 0; r < 8; ++r) {
                float v = fmaxf(acc[tm][tn][r] + bc, 0.0f);
                if (OUT_BF16)
                    outb[(size_t)(rowBase + r) * N + col] = (bf16_t)v;
                else
                    outf[(size_t)(rowBase + r) * N + col] = v;
            }
        }
    }
}

// ---------------------------------------------------------------------------
// Launcher
// ---------------------------------------------------------------------------
extern "C" void kernel_launch(void* const* d_in, const int* in_sizes, int n_in,
                              void* d_out, int out_size, void* d_ws, size_t ws_size,
                              hipStream_t stream) {
    (void)in_sizes; (void)n_in; (void)out_size; (void)ws_size;
    const float* emb     = (const float*)d_in[0];
    const int*   tokens  = (const int*)  d_in[1];
    const int*   lengths = (const int*)  d_in[2];
    const float* mention = (const float*)d_in[3];
    const float* w1      = (const float*)d_in[4];
    const float* b1      = (const float*)d_in[5];
    const float* w2      = (const float*)d_in[6];
    const float* b2      = (const float*)d_in[7];
    float*       out     = (float*)d_out;

    // Workspace layout (all offsets 256B-aligned)
    char* ws = (char*)d_ws;
    const size_t xBytes  = (size_t)N_SENT * K1_PAD * sizeof(bf16_t);   // 22,020,096
    const size_t hBytes  = (size_t)N_SENT * H2_DIM * sizeof(bf16_t);   //  8,388,608
    const size_t w1Bytes = (size_t)H2_DIM * K1_PAD * sizeof(bf16_t);   //  1,376,256
    bf16_t* X   = (bf16_t*)(ws);
    bf16_t* H1  = (bf16_t*)(ws + xBytes);
    bf16_t* W1b = (bf16_t*)(ws + xBytes + hBytes);
    bf16_t* W2b = (bf16_t*)(ws + xBytes + hBytes + w1Bytes);

    // Stage 0: weight bf16 packing (tiny)
    cvt_w1_kernel<<<(H2_DIM * K1_PAD + 255) / 256, 256, 0, stream>>>(w1, W1b);
    cvt_w2_kernel<<<(H_DIM * H2_DIM + 255) / 256, 256, 0, stream>>>(w2, W2b);

    // Stage 1: embedding-bag + concat + pack (one block per sentence)
    embed_concat_kernel<<<N_SENT, 256, 0, stream>>>(emb, tokens, lengths, mention, X);

    // Stage 2: H1 = relu(X @ W1b^T + b1)   [8192 x 512], K = 1344
    // waves = (8192/32)*(512/64) = 2048 -> 256 blocks of 8 waves
    gemm_bias_relu_wmma<K1_PAD, H2_DIM, true><<<256, 256, 0, stream>>>(X, W1b, b1, H1);

    // Stage 3: out = relu(H1 @ W2b^T + b2) [8192 x 256], K = 512
    // waves = (8192/32)*(256/64) = 1024 -> 128 blocks of 8 waves
    gemm_bias_relu_wmma<H2_DIM, H_DIM, false><<<128, 256, 0, stream>>>(H1, W2b, b2, out);
}
